// Block_14070312862412
// MI455X (gfx1250) — matrix operand order repaired and checked by reading, not compiled, not run on hardware
//
#include <hip/hip_runtime.h>
#include <hip/hip_bf16.h>
#include <math.h>

typedef __bf16 bf16_t;
typedef __attribute__((ext_vector_type(16))) __bf16 v16bf;
typedef __attribute__((ext_vector_type(8)))  __bf16 bf16x8;
typedef __attribute__((ext_vector_type(8)))  float  v8f;

#define EMBED_DIM 512
#define NUM_HEADS 8
#define HEAD_DIM 64
#define NUM_EXPERTS 16
#define TOPK 8
#define BATCH 2
#define SEQ 1024
#define NTOK (BATCH*SEQ)
#define HID 2048

#define BM 128
#define BN 128
#define BK 32
#define LDK 48   // padded LDS row stride (96B: 16B-aligned, bank-staggered)

enum { EPI_BIAS = 0, EPI_RESID = 1, EPI_GATE = 2 };

// ---------------------------------------------------------------------------
// bf16 WMMA GEMM, software-pipelined with double-buffered LDS.
// C[M,N] = epilogue(A[M,K] (fp32) * B[K,N] (fp32)); tiles converted to bf16
// on the fly; 8 waves in a 4(M) x 2(N) grid, each wave owns a 32x64 block
// (2x4 v_wmma_f32_16x16x32_bf16 tiles, f32 accumulation).
// ---------------------------------------------------------------------------
template <int EPI>
__global__ __launch_bounds__(256)
void gemm_bf16_wmma(const float* __restrict__ A, const float* __restrict__ Bw,
                    const float* __restrict__ bias, float* __restrict__ C,
                    int M, int N, int K,
                    const float* __restrict__ resid,
                    const float* __restrict__ gates, int expert)
{
    __shared__ __align__(16) bf16_t lA[2][BM][LDK];
    __shared__ __align__(16) bf16_t lB[2][BN][LDK];

    const int tid  = threadIdx.x;
    const int lane = tid & 31;
    const int wave = tid >> 5;      // 0..7
    const int wm   = wave & 3;      // M dir (4)
    const int wn   = wave >> 2;     // N dir (2)
    const int half = lane >> 4;
    const int l16  = lane & 15;
    const int blockN = blockIdx.x * BN;
    const int blockM = blockIdx.y * BM;

    // staging task decomposition (4 tasks per thread, constant indices)
    int aRow[4], aCol[4], bN[4], bK[4];
    #pragma unroll
    for (int i = 0; i < 4; ++i) {
        int f = tid + 256 * i;
        aRow[i] = f >> 3;          // A: 128 rows x 8 float4 along k
        aCol[i] = (f & 7) << 2;
        bN[i]   = f & 127;         // B: 128 n-cols x 8 groups of 4 k
        bK[i]   = (f >> 7) << 2;
    }

    const float* Abase = A + (size_t)blockM * K;
    const float* Bbase = Bw + blockN;

    float4 pa[4];
    float  pb[4][4];

    // ---- prefetch + stage k-block 0 into buffer 0 ----
    #pragma unroll
    for (int i = 0; i < 4; ++i)
        pa[i] = *(const float4*)(Abase + (size_t)aRow[i] * K + aCol[i]);
    #pragma unroll
    for (int i = 0; i < 4; ++i)
        #pragma unroll
        for (int q = 0; q < 4; ++q)
            pb[i][q] = Bbase[(size_t)(bK[i] + q) * N + bN[i]];
    #pragma unroll
    for (int i = 0; i < 4; ++i) {
        lA[0][aRow[i]][aCol[i] + 0] = (bf16_t)pa[i].x;
        lA[0][aRow[i]][aCol[i] + 1] = (bf16_t)pa[i].y;
        lA[0][aRow[i]][aCol[i] + 2] = (bf16_t)pa[i].z;
        lA[0][aRow[i]][aCol[i] + 3] = (bf16_t)pa[i].w;
        lB[0][bN[i]][bK[i] + 0] = (bf16_t)pb[i][0];
        lB[0][bN[i]][bK[i] + 1] = (bf16_t)pb[i][1];
        lB[0][bN[i]][bK[i] + 2] = (bf16_t)pb[i][2];
        lB[0][bN[i]][bK[i] + 3] = (bf16_t)pb[i][3];
    }
    __syncthreads();

    v8f acc[2][4] = {};
    const int nk = K / BK;
    for (int t = 0; t < nk; ++t) {
        const int buf = t & 1;
        const int kb  = (t + 1) * BK;
        // prefetch next k-block into registers (latency hides under WMMAs)
        if (t + 1 < nk) {
            #pragma unroll
            for (int i = 0; i < 4; ++i)
                pa[i] = *(const float4*)(Abase + (size_t)aRow[i] * K + kb + aCol[i]);
            #pragma unroll
            for (int i = 0; i < 4; ++i)
                #pragma unroll
                for (int q = 0; q < 4; ++q)
                    pb[i][q] = Bbase[(size_t)(kb + bK[i] + q) * N + bN[i]];
        }
        // fragments per documented gfx1250 layouts
        v16bf afr[2], bfr[4];
        #pragma unroll
        for (int i = 0; i < 2; ++i) {
            const int row = wm * 32 + i * 16 + l16;
            bf16x8 lo = *(const bf16x8*)&lA[buf][row][8 * half];
            bf16x8 hi = *(const bf16x8*)&lA[buf][row][16 + 8 * half];
            #pragma unroll
            for (int q = 0; q < 8; ++q) { afr[i][q] = lo[q]; afr[i][8 + q] = hi[q]; }
        }
        #pragma unroll
        for (int j = 0; j < 4; ++j) {
            const int col = wn * 64 + j * 16 + l16;
            bf16x8 lo = *(const bf16x8*)&lB[buf][col][8 * half];
            bf16x8 hi = *(const bf16x8*)&lB[buf][col][16 + 8 * half];
            #pragma unroll
            for (int q = 0; q < 8; ++q) { bfr[j][q] = lo[q]; bfr[j][8 + q] = hi[q]; }
        }
        #pragma unroll
        for (int i = 0; i < 2; ++i)
            #pragma unroll
            for (int j = 0; j < 4; ++j)
                acc[i][j] = __builtin_amdgcn_wmma_f32_16x16x32_bf16(
                    false, afr[i], false, bfr[j], (short)0, acc[i][j], false, false);
        // stage prefetched block into the other buffer
        if (t + 1 < nk) {
            const int nb = buf ^ 1;
            #pragma unroll
            for (int i = 0; i < 4; ++i) {
                lA[nb][aRow[i]][aCol[i] + 0] = (bf16_t)pa[i].x;
                lA[nb][aRow[i]][aCol[i] + 1] = (bf16_t)pa[i].y;
                lA[nb][aRow[i]][aCol[i] + 2] = (bf16_t)pa[i].z;
                lA[nb][aRow[i]][aCol[i] + 3] = (bf16_t)pa[i].w;
                lB[nb][bN[i]][bK[i] + 0] = (bf16_t)pb[i][0];
                lB[nb][bN[i]][bK[i] + 1] = (bf16_t)pb[i][1];
                lB[nb][bN[i]][bK[i] + 2] = (bf16_t)pb[i][2];
                lB[nb][bN[i]][bK[i] + 3] = (bf16_t)pb[i][3];
            }
        }
        __syncthreads();
    }

    // ---- epilogue: D rows = v + 8*half (consecutive), pointer-stepped ----
    #pragma unroll
    for (int i = 0; i < 2; ++i) {
        const int row0 = blockM + wm * 32 + i * 16 + 8 * half;
        #pragma unroll
        for (int j = 0; j < 4; ++j) {
            const int col = blockN + wn * 64 + j * 16 + l16;
            const float bv = bias[col];
            float* cp = C + (size_t)row0 * N + col;
            if (EPI == EPI_BIAS) {
                #pragma unroll
                for (int v = 0; v < 8; ++v) { *cp = acc[i][j][v] + bv; cp += N; }
            } else if (EPI == EPI_RESID) {
                const float* rp = resid + (size_t)row0 * N + col;
                #pragma unroll
                for (int v = 0; v < 8; ++v) { *cp = *rp + acc[i][j][v] + bv; cp += N; rp += N; }
            } else { // EPI_GATE: out += gate[row,e] * (acc + bias)
                const float* gp = gates + (size_t)row0 * NUM_EXPERTS + expert;
                #pragma unroll
                for (int v = 0; v < 8; ++v) {
                    *cp += *gp * (acc[i][j][v] + bv);
                    cp += N; gp += NUM_EXPERTS;
                }
            }
        }
    }
}

// ---------------------------------------------------------------------------
// RMSNorm: one wave per token
// ---------------------------------------------------------------------------
__global__ __launch_bounds__(256)
void rmsnorm_kernel(const float* __restrict__ x, const float* __restrict__ g,
                    const float* __restrict__ b, float* __restrict__ xn)
{
    int wave = threadIdx.x >> 5, lane = threadIdx.x & 31;
    int n = blockIdx.x * 8 + wave;
    const float* xr = x + (size_t)n * EMBED_DIM;
    float v[16]; float ss = 0.f;
    #pragma unroll
    for (int i = 0; i < 16; ++i) { v[i] = xr[lane + 32 * i]; ss += v[i] * v[i]; }
    #pragma unroll
    for (int m = 16; m; m >>= 1) ss += __shfl_xor(ss, m, 32);
    float inv = rsqrtf(ss * (1.f / EMBED_DIM) + 1e-6f);
    float* o = xn + (size_t)n * EMBED_DIM;
    #pragma unroll
    for (int i = 0; i < 16; ++i) { int c = lane + 32 * i; o[c] = v[i] * inv * g[c] + b[c]; }
}

// ---------------------------------------------------------------------------
// QK-normalized causal attention: one wave per query row, two-pass softmax
// score = alpha[h] * (q.k) / ((|q|+eps)(|k|+eps))   (sqrt(hd) factors cancel)
// ---------------------------------------------------------------------------
__global__ __launch_bounds__(256)
void attention_kernel(const float* __restrict__ qkv, const float* __restrict__ alpha,
                      float* __restrict__ y)
{
    __shared__ float sc[8][SEQ];
    int wave = threadIdx.x >> 5, lane = threadIdx.x & 31;
    int gw = blockIdx.x * 8 + wave;
    int t  = gw % SEQ;
    int bh = gw / SEQ;
    int h  = bh % NUM_HEADS;
    int b  = bh / NUM_HEADS;
    int n  = b * SEQ + t;

    const float* qp = qkv + (size_t)n * (3 * EMBED_DIM) + h * HEAD_DIM;
    float qa = qp[2 * lane], qb = qp[2 * lane + 1];
    float qs = qa * qa + qb * qb;
    #pragma unroll
    for (int m = 16; m; m >>= 1) qs += __shfl_xor(qs, m, 32);
    float qn = sqrtf(qs) + 1e-5f;
    float ah = alpha[h];

    const float* kbase = qkv + (size_t)b * SEQ * (3 * EMBED_DIM) + EMBED_DIM + h * HEAD_DIM;
    float mx = -1e30f;
    for (int k = 0; k <= t; ++k) {
        const float* kp = kbase + (size_t)k * (3 * EMBED_DIM);
        float ka = kp[2 * lane], kb2 = kp[2 * lane + 1];
        float d  = qa * ka + qb * kb2;
        float s2 = ka * ka + kb2 * kb2;
        #pragma unroll
        for (int m = 16; m; m >>= 1) { d += __shfl_xor(d, m, 32); s2 += __shfl_xor(s2, m, 32); }
        float sv = ah * d / (qn * (sqrtf(s2) + 1e-5f));
        if (lane == 0) sc[wave][k] = sv;
        mx = fmaxf(mx, sv);
    }
    float sum = 0.f, y0 = 0.f, y1 = 0.f;
    const float* vbase = qkv + (size_t)b * SEQ * (3 * EMBED_DIM) + 2 * EMBED_DIM + h * HEAD_DIM;
    for (int k = 0; k <= t; ++k) {
        float p = __expf(sc[wave][k] - mx);
        const float* vp = vbase + (size_t)k * (3 * EMBED_DIM);
        sum += p; y0 += p * vp[2 * lane]; y1 += p * vp[2 * lane + 1];
    }
    float* yp = y + (size_t)n * EMBED_DIM + h * HEAD_DIM;
    yp[2 * lane] = y0 / sum;
    yp[2 * lane + 1] = y1 / sum;
}

// ---------------------------------------------------------------------------
// Router: softmax over 16 experts + top-8 normalized gates; one wave/token
// ---------------------------------------------------------------------------
__global__ __launch_bounds__(256)
void router_kernel(const float* __restrict__ x1, const float* __restrict__ rw,
                   const float* __restrict__ rb, float* __restrict__ gates)
{
    int wave = threadIdx.x >> 5, lane = threadIdx.x & 31;
    int n = blockIdx.x * 8 + wave;
    const float* xr = x1 + (size_t)n * EMBED_DIM;
    float lg[16];
    #pragma unroll
    for (int e = 0; e < 16; ++e) lg[e] = 0.f;
    for (int i = 0; i < 16; ++i) {
        int c = lane + 32 * i;
        float xv = xr[c];
        const float4* wr = (const float4*)(rw + (size_t)c * 16);
        #pragma unroll
        for (int q = 0; q < 4; ++q) {
            float4 w4 = wr[q];
            lg[q * 4 + 0] += xv * w4.x; lg[q * 4 + 1] += xv * w4.y;
            lg[q * 4 + 2] += xv * w4.z; lg[q * 4 + 3] += xv * w4.w;
        }
    }
    #pragma unroll
    for (int e = 0; e < 16; ++e)
        #pragma unroll
        for (int m = 16; m; m >>= 1) lg[e] += __shfl_xor(lg[e], m, 32);
    float mx = -1e30f;
    #pragma unroll
    for (int e = 0; e < 16; ++e) { lg[e] += rb[e]; mx = fmaxf(mx, lg[e]); }
    float s = 0.f;
    #pragma unroll
    for (int e = 0; e < 16; ++e) { lg[e] = __expf(lg[e] - mx); s += lg[e]; }
    #pragma unroll
    for (int e = 0; e < 16; ++e) lg[e] /= s;
    unsigned sel = 0; float ts = 0.f;
    for (int it = 0; it < TOPK; ++it) {
        int best = 0; float bv = -1.f;
        #pragma unroll
        for (int e = 0; e < 16; ++e)
            if (!((sel >> e) & 1) && lg[e] > bv) { bv = lg[e]; best = e; }
        sel |= 1u << best; ts += bv;
    }
    if (lane < 16)
        gates[(size_t)n * NUM_EXPERTS + lane] = ((sel >> lane) & 1) ? lg[lane] / ts : 0.f;
}

// ---------------------------------------------------------------------------
// SwiGLU: s[n,j] = silu(c[n, HID+j]) * c[n, j]
// ---------------------------------------------------------------------------
__global__ __launch_bounds__(256)
void swiglu_kernel(const float* __restrict__ c, float* __restrict__ s)
{
    size_t idx = (size_t)blockIdx.x * 256 + threadIdx.x;
    size_t n = idx / HID, j = idx - n * HID;
    float a  = c[n * (2 * HID) + j];
    float bb = c[n * (2 * HID) + HID + j];
    s[idx] = (bb / (1.f + __expf(-bb))) * a;
}

__global__ __launch_bounds__(256)
void copy_kernel(const float* __restrict__ src, float* __restrict__ dst)
{
    size_t i = (size_t)blockIdx.x * 256 + threadIdx.x;
    ((float4*)dst)[i] = ((const float4*)src)[i];
}

// ---------------------------------------------------------------------------
extern "C" void kernel_launch(void* const* d_in, const int* in_sizes, int n_in,
                              void* d_out, int out_size, void* d_ws, size_t ws_size,
                              hipStream_t stream)
{
    (void)in_sizes; (void)n_in; (void)out_size; (void)ws_size;
    const float* x        = (const float*)d_in[0];
    const float* g        = (const float*)d_in[1];
    const float* bb       = (const float*)d_in[2];
    const float* c_attn_w = (const float*)d_in[3];
    const float* c_attn_b = (const float*)d_in[4];
    const float* alpha    = (const float*)d_in[5];
    const float* c_proj_w = (const float*)d_in[6];
    const float* c_proj_b = (const float*)d_in[7];
    const float* router_w = (const float*)d_in[8];
    const float* router_b = (const float*)d_in[9];
    const float* w_in     = (const float*)d_in[10];
    const float* b_in     = (const float*)d_in[11];
    const float* w1       = (const float*)d_in[12];
    const float* b1       = (const float*)d_in[13];
    const float* w2       = (const float*)d_in[14];
    const float* b2       = (const float*)d_in[15];
    const float* w_out    = (const float*)d_in[16];
    const float* b_out    = (const float*)d_in[17];
    float* out = (float*)d_out;

    char* wsb = (char*)d_ws;
    const size_t MB = 1024 * 1024;
    float* xn    = (float*)(wsb);              // 4 MB  [NTOK,512]
    float* qkv   = (float*)(wsb + 4  * MB);    // 12 MB [NTOK,1536]
    float* attny = (float*)(wsb + 16 * MB);    // 4 MB  [NTOK,512]
    float* x1    = (float*)(wsb + 20 * MB);    // 4 MB  [NTOK,512]
    float* gates = (float*)(wsb + 24 * MB);    // 128KB [NTOK,16]
    float* hbuf  = (float*)(wsb + 25 * MB);    // 16 MB [NTOK,2048]
    float* cbuf  = (float*)(wsb + 41 * MB);    // 32 MB [NTOK,4096]
    float* sbuf  = (float*)(wsb + 73 * MB);    // 16 MB [NTOK,2048]
    float* obuf  = (float*)(wsb + 89 * MB);    // 16 MB [NTOK,2048]

    rmsnorm_kernel<<<NTOK / 8, 256, 0, stream>>>(x, g, bb, xn);

    gemm_bf16_wmma<EPI_BIAS><<<dim3(3 * EMBED_DIM / BN, NTOK / BM), 256, 0, stream>>>(
        xn, c_attn_w, c_attn_b, qkv, NTOK, 3 * EMBED_DIM, EMBED_DIM, nullptr, nullptr, 0);

    attention_kernel<<<BATCH * NUM_HEADS * SEQ / 8, 256, 0, stream>>>(qkv, alpha, attny);

    gemm_bf16_wmma<EPI_RESID><<<dim3(EMBED_DIM / BN, NTOK / BM), 256, 0, stream>>>(
        attny, c_proj_w, c_proj_b, x1, NTOK, EMBED_DIM, EMBED_DIM, x, nullptr, 0);

    router_kernel<<<NTOK / 8, 256, 0, stream>>>(x1, router_w, router_b, gates);

    copy_kernel<<<NTOK * EMBED_DIM / 4 / 256, 256, 0, stream>>>(x1, out);

    for (int e = 0; e < NUM_EXPERTS; ++e) {
        gemm_bf16_wmma<EPI_BIAS><<<dim3(HID / BN, NTOK / BM), 256, 0, stream>>>(
            x1, w_in + (size_t)e * EMBED_DIM * HID, b_in + (size_t)e * HID, hbuf,
            NTOK, HID, EMBED_DIM, nullptr, nullptr, 0);
        gemm_bf16_wmma<EPI_BIAS><<<dim3(2 * HID / BN, NTOK / BM), 256, 0, stream>>>(
            hbuf, w1 + (size_t)e * HID * 2 * HID, b1 + (size_t)e * 2 * HID, cbuf,
            NTOK, 2 * HID, HID, nullptr, nullptr, 0);
        swiglu_kernel<<<NTOK * HID / 256, 256, 0, stream>>>(cbuf, sbuf);
        gemm_bf16_wmma<EPI_BIAS><<<dim3(HID / BN, NTOK / BM), 256, 0, stream>>>(
            sbuf, w2 + (size_t)e * HID * HID, b2 + (size_t)e * HID, obuf,
            NTOK, HID, HID, nullptr, nullptr, 0);
        gemm_bf16_wmma<EPI_GATE><<<dim3(EMBED_DIM / BN, NTOK / BM), 256, 0, stream>>>(
            obuf, w_out + (size_t)e * HID * EMBED_DIM, b_out + (size_t)e * EMBED_DIM, out,
            NTOK, EMBED_DIM, HID, nullptr, gates, e);
    }
}
